// AtariNet_75307956568738
// MI455X (gfx1250) — matrix operand order repaired and checked by reading, not compiled, not run on hardware
//
#include <hip/hip_runtime.h>
#include <hip/hip_bf16.h>
#include <math.h>

typedef _Float16 f16;
typedef __attribute__((ext_vector_type(16))) _Float16 v16h;
typedef __attribute__((ext_vector_type(8)))  _Float16 v8h;
typedef __attribute__((ext_vector_type(8)))  float    v8f;

#define WMMA_F16(a,b,c) __builtin_amdgcn_wmma_f32_16x16x32_f16(false,(a),false,(b),(short)0,(c),false,false)

// ---------------- utility / packing kernels ----------------

__global__ void frames_to_half_k(const int* __restrict__ frame, f16* __restrict__ out,
                                 int frame0, int nframes)
{
  size_t total = (size_t)nframes * 64 * 64 * 3;
  size_t i = (size_t)blockIdx.x * blockDim.x + threadIdx.x;
  if (i >= total) return;
  int c = (int)(i % 3);
  size_t r = i / 3;
  int x = (int)(r % 64); r /= 64;
  int y = (int)(r % 64);
  int fr = (int)(r / 64);
  int g = frame[(((size_t)(frame0 + fr) * 3 + c) * 64 + y) * 64 + x];
  out[i] = (f16)((float)g * (1.0f / 255.0f));
}

__global__ void cast_f32_f16_k(const float* __restrict__ in, f16* __restrict__ out, size_t n)
{
  size_t i = (size_t)blockIdx.x * blockDim.x + threadIdx.x;
  if (i < n) out[i] = (f16)in[i];
}

// OIHW float -> [O][(ky*3+kx)*Cin + c] f16
__global__ void pack_conv_w_k(const float* __restrict__ w, f16* __restrict__ out,
                              int Cout, int Cin)
{
  int K = 9 * Cin;
  int i = blockIdx.x * blockDim.x + threadIdx.x;
  if (i >= Cout * K) return;
  int m = i / K, k = i - m * K;
  int c = k % Cin;
  int kyx = k / Cin;
  int ky = kyx / 3, kx = kyx - ky * 3;
  out[i] = (f16)w[((m * Cin + c) * 3 + ky) * 3 + kx];
}

// Pack [Wih | Whh] -> f16 [1088][544], bias sum -> f32 [1088]
__global__ void pack_lstm_k(const float* __restrict__ Wih, const float* __restrict__ Whh,
                            const float* __restrict__ bih, const float* __restrict__ bhh,
                            f16* __restrict__ Wcat, float* __restrict__ bsum)
{
  int i = blockIdx.x * blockDim.x + threadIdx.x;
  if (i >= 1088 * 544) return;
  int n = i / 544, k = i - n * 544;
  f16 v = (f16)0;
  if (n < 1076) {
    if (k < 269)      v = (f16)Wih[n * 269 + k];
    else if (k < 538) v = (f16)Whh[n * 269 + (k - 269)];
  }
  Wcat[i] = v;
  if (k == 0) bsum[n] = (n < 1076) ? (bih[n] + bhh[n]) : 0.0f;
}

// channels-last [2048][8][8][32] -> NCHW-flat [2048][2048] (k = c*64 + y*8 + x)
__global__ void repack_nchw_k(const f16* __restrict__ in, f16* __restrict__ out)
{
  size_t i = (size_t)blockIdx.x * blockDim.x + threadIdx.x;
  if (i >= (size_t)2048 * 2048) return;
  int row = (int)(i >> 11), k = (int)(i & 2047);
  int c = k >> 6, rem = k & 63, y = rem >> 3, x = rem & 7;
  out[i] = in[(((size_t)row * 8 + y) * 8 + x) * 32 + c];
}

// ---------------- implicit-GEMM 3x3 SAME conv via WMMA ----------------
// in  : [nframes][H][W][CIN]  f16 (channels-last)
// wk  : [COUT][9*CIN]         f16, k = (ky*3+kx)*CIN + c
// out : [nframes][H][W][COUT] f16 ;  out = conv(reluIn? relu(in):in) + bias (+resid)
// NPT pixel-tiles per wave reuse the A (weight) fragment across NPT WMMAs.
template <int CIN, int COUT, int NPT>
__global__ void __launch_bounds__(32) conv3x3_wmma_k(
    const f16* __restrict__ in, const f16* __restrict__ wk,
    const float* __restrict__ bias, const f16* __restrict__ resid,
    f16* __restrict__ out, int H, int W, int reluIn)
{
  constexpr int KDIM = 9 * CIN;
  constexpr int KT = (KDIM + 31) / 32;
  constexpr bool FAST = (CIN % 16 == 0);     // contiguous-K fragment path
  const int lane = threadIdx.x;
  const int tn = blockIdx.x;
  const int tm = blockIdx.y;
  const int HW = H * W;

  const int rl  = lane & 15;
  const int hiA = (lane >= 16) ? 8 : 0;
  const int hiB = (lane >= 16) ? 16 : 0;

  int p[NPT], fr[NPT], y[NPT], x[NPT];
#pragma unroll
  for (int j = 0; j < NPT; ++j) {
    p[j] = (tn * NPT + j) * 16 + rl;
    fr[j] = p[j] / HW;
    int rem = p[j] - fr[j] * HW;
    y[j] = rem / W;
    x[j] = rem - y[j] * W;
  }
  const int mrow = tm * 16 + rl;

  v8f acc[NPT];
#pragma unroll
  for (int j = 0; j < NPT; ++j) acc[j] = (v8f){};

  for (int kt = 0; kt < KT; ++kt) {
    const int kbase = kt * 32;
    // ---- A fragment: two contiguous 8-half chunks per lane ----
    v16h av = {};
    if constexpr (FAST) {
      const f16* wrow = wk + (size_t)mrow * KDIM;
      int ka0 = kbase + hiB;
      int ka1 = kbase + hiB + 8;
      v8h a0 = (ka0 < KDIM) ? *(const v8h*)(wrow + ka0) : (v8h){};
      v8h a1 = (ka1 < KDIM) ? *(const v8h*)(wrow + ka1) : (v8h){};
#pragma unroll
      for (int e = 0; e < 8; ++e) { av[e] = a0[e]; av[e + 8] = a1[e]; }
    } else {
#pragma unroll
      for (int vv = 0; vv < 8; ++vv)
#pragma unroll
        for (int hh = 0; hh < 2; ++hh) {
          int ka = kbase + hiB + vv * 2 + hh;
          f16 aval = (f16)0;
          if (ka < KDIM) aval = wk[(size_t)mrow * KDIM + ka];
          av[vv * 2 + hh] = aval;
        }
    }
    // ---- B fragment(s): one contiguous 16-half chunk per lane (FAST) ----
#pragma unroll
    for (int j = 0; j < NPT; ++j) {
      v16h bv = {};
      if constexpr (FAST) {
        int kstart = kbase + hiB;              // 16 consecutive K values
        int tap = kstart / CIN;                // constant-folded
        int c0 = kstart - tap * CIN;
        if (tap < 9) {
          int ky = tap / 3, kx = tap - ky * 3;
          int yy = y[j] + ky - 1, xx = x[j] + kx - 1;
          if ((unsigned)yy < (unsigned)H && (unsigned)xx < (unsigned)W) {
            const f16* src = in + (((size_t)fr[j] * H + yy) * W + xx) * CIN + c0;
            bv = *(const v16h*)src;
            if (reluIn) {
#pragma unroll
              for (int e = 0; e < 16; ++e) bv[e] = (bv[e] < (f16)0) ? (f16)0 : bv[e];
            }
          }
        }
      } else {
#pragma unroll
        for (int vv = 0; vv < 8; ++vv)
#pragma unroll
          for (int hh = 0; hh < 2; ++hh) {
            int kb = kbase + hiB + vv * 2 + hh;
            f16 bval = (f16)0;
            if (kb < KDIM) {
              int kyx = kb / CIN;
              int c = kb - kyx * CIN;
              int ky = kyx / 3, kx = kyx - ky * 3;
              int yy = y[j] + ky - 1, xx = x[j] + kx - 1;
              if ((unsigned)yy < (unsigned)H && (unsigned)xx < (unsigned)W) {
                f16 t = in[(((size_t)fr[j] * H + yy) * W + xx) * CIN + c];
                if (reluIn && t < (f16)0) t = (f16)0;
                bval = t;
              }
            }
            bv[vv * 2 + hh] = bval;
          }
      }
      acc[j] = WMMA_F16(av, bv, acc[j]);
    }
  }

  // ---- store: D rows are contiguous output channels -> one b128 per subtile ----
  const int m0 = tm * 16 + hiA;
#pragma unroll
  for (int j = 0; j < NPT; ++j) {
    f16* dst = out + (size_t)p[j] * COUT + m0;
    v8h st;
    if (resid) {
      v8h rv = *(const v8h*)(resid + (size_t)p[j] * COUT + m0);
#pragma unroll
      for (int d = 0; d < 8; ++d) st[d] = (f16)(acc[j][d] + bias[m0 + d] + (float)rv[d]);
    } else {
#pragma unroll
      for (int d = 0; d < 8; ++d) st[d] = (f16)(acc[j][d] + bias[m0 + d]);
    }
    *(v8h*)dst = st;
  }
}

// ---------------- 3x3 stride-2 maxpool, SAME pad (1,1) ----------------
__global__ void maxpool3x3s2_k(const f16* __restrict__ in, f16* __restrict__ out,
                               int nframes, int H, int W, int C)
{
  int HO = H / 2, WO = W / 2;
  size_t total = (size_t)nframes * HO * WO * C;
  size_t i = (size_t)blockIdx.x * blockDim.x + threadIdx.x;
  if (i >= total) return;
  int c = (int)(i % C);
  size_t r = i / C;
  int xo = (int)(r % WO); r /= WO;
  int yo = (int)(r % HO);
  int fr = (int)(r / HO);
  float m = -3.402823466e38f;
  for (int dy = -1; dy <= 1; ++dy)
    for (int dx = -1; dx <= 1; ++dx) {
      int yy = 2 * yo + dy, xx = 2 * xo + dx;
      if ((unsigned)yy < (unsigned)H && (unsigned)xx < (unsigned)W) {
        float v = (float)in[(((size_t)fr * H + yy) * W + xx) * C + c];
        m = fmaxf(m, v);
      }
    }
  out[i] = (f16)m;
}

// ---------------- FC 256x2048x2048 via WMMA (A = weights, B = activations) ----
// actN: [2048 frames][2048] f16, NCHW-flat K; wk: [256][2048] f16.
// D: M = feature (contiguous store), N = frame. 4 frame-tiles per wave.
__global__ void __launch_bounds__(32) fc_wmma_k(
    const f16* __restrict__ actN, const f16* __restrict__ wk,
    const float* __restrict__ bias, f16* __restrict__ out)
{
  const int lane = threadIdx.x;
  const int tmM = blockIdx.x;        // 0..15 feature tiles (256)
  const int tn0 = blockIdx.y * 4;    // frame tiles (128 total), 4 per wave
  const int rl  = lane & 15;
  const int hiA = (lane >= 16) ? 8 : 0;
  const int hiB = (lane >= 16) ? 16 : 0;
  const int mrow = tmM * 16 + rl;

  int frameN[4];
#pragma unroll
  for (int j = 0; j < 4; ++j) frameN[j] = (tn0 + j) * 16 + rl;

  v8f acc[4];
#pragma unroll
  for (int j = 0; j < 4; ++j) acc[j] = (v8f){};

  for (int kt = 0; kt < 64; ++kt) {
    const int kbase = kt * 32;
    v16h av;
    {
      const f16* wrow = wk + (size_t)mrow * 2048;
      v8h a0 = *(const v8h*)(wrow + kbase + hiB);
      v8h a1 = *(const v8h*)(wrow + kbase + hiB + 8);
#pragma unroll
      for (int e = 0; e < 8; ++e) { av[e] = a0[e]; av[e + 8] = a1[e]; }
    }
#pragma unroll
    for (int j = 0; j < 4; ++j) {
      v16h bv = *(const v16h*)(actN + (size_t)frameN[j] * 2048 + kbase + hiB);
      acc[j] = WMMA_F16(av, bv, acc[j]);
    }
  }
  const int m0 = tmM * 16 + hiA;
#pragma unroll
  for (int j = 0; j < 4; ++j) {
    v8h st;
#pragma unroll
    for (int d = 0; d < 8; ++d) {
      float v = acc[j][d] + bias[m0 + d];
      st[d] = (f16)((v < 0.0f) ? 0.0f : v);
    }
    *(v8h*)(out + (size_t)frameN[j] * 256 + m0) = st;
  }
}

// ---------------- core input concat: [2048][288] f16 (269 used) ----------------
__global__ void build_core_k(const f16* __restrict__ X, const float* __restrict__ reward,
                             const int* __restrict__ last_action,
                             const float* __restrict__ apos, const float* __restrict__ adir,
                             const float* __restrict__ tpos, f16* __restrict__ ci)
{
  size_t i = (size_t)blockIdx.x * blockDim.x + threadIdx.x;
  if (i >= (size_t)2048 * 288) return;
  int row = (int)(i / 288), j = (int)(i % 288);
  f16 v = (f16)0;
  if (j < 256)       v = X[(size_t)row * 256 + j];
  else if (j == 256) { float r = reward[row]; v = (f16)fminf(1.0f, fmaxf(-1.0f, r)); }
  else if (j < 263)  v = (f16)((last_action[row] == (j - 257)) ? 1.0f : 0.0f);
  else if (j < 265)  v = (f16)apos[row * 2 + (j - 263)];
  else if (j < 267)  v = (f16)adir[row * 2 + (j - 265)];
  else if (j < 269)  v = (f16)tpos[row * 2 + (j - 267)];
  ci[i] = v;
}

// ---------------- persistent 2-layer LSTM, single workgroup, WMMA ----------------
// GEMM roles: A = Wcat rows (M = gate outputs), B = z columns (N = batch, from LDS).
__global__ void __launch_bounds__(1024) lstm_seq_k(
    const f16* __restrict__ ci,          // [64][32][288]
    const unsigned char* __restrict__ done,
    const float* __restrict__ h0, const float* __restrict__ c0,  // [2][32][269]
    const f16* __restrict__ Wcat,        // [2][1088][544]
    const float* __restrict__ bsum,      // [2][1088]
    float* __restrict__ hstate, float* __restrict__ cstate,      // [2][32][272]
    float* __restrict__ gates,           // [32][1088]
    float* __restrict__ outs)            // [64][32][269]
{
  __shared__ f16 z[32][544];
  __shared__ float nd[32];
  const int tid = threadIdx.x;
  const int lane = tid & 31;
  const int wave = tid >> 5;

  for (int i = tid; i < 2 * 32 * 272; i += 1024) {
    int k = i % 272; int b = (i / 272) % 32; int l = i / (272 * 32);
    float hv = 0.0f, cv = 0.0f;
    if (k < 269) { hv = h0[(l * 32 + b) * 269 + k]; cv = c0[(l * 32 + b) * 269 + k]; }
    hstate[i] = hv; cstate[i] = cv;
  }
  __syncthreads();

  for (int t = 0; t < 64; ++t) {
    if (tid < 32) nd[tid] = done[t * 32 + tid] ? 0.0f : 1.0f;
    __syncthreads();
    for (int l = 0; l < 2; ++l) {
      // z = [inp(269), h_l * nd (269), pad] in LDS
      for (int i = tid; i < 32 * 544; i += 1024) {
        int b = i / 544, k = i % 544;
        f16 v = (f16)0;
        if (k < 269)
          v = (l == 0) ? ci[((size_t)t * 32 + b) * 288 + k]
                       : (f16)hstate[(0 * 32 + b) * 272 + k];
        else if (k < 538)
          v = (f16)(hstate[(l * 32 + b) * 272 + (k - 269)] * nd[b]);
        z[b][k] = v;
      }
      __syncthreads();

      const f16* W = Wcat + (size_t)l * 1088 * 544;
      const int rl  = lane & 15;
      const int hiA = (lane >= 16) ? 8 : 0;
      const int hiB = (lane >= 16) ? 16 : 0;
      // 68 gate-tiles x 2 batch-tiles = 136 jobs over 32 waves
      for (int job = wave; job < 136; job += 32) {
        int gt = job >> 1;        // gate tile
        int bt = job & 1;         // batch tile
        int arow = gt * 16 + rl;  // gate row (A)
        int bcol = bt * 16 + rl;  // batch column (B)
        v8f acc = {};
        for (int kt = 0; kt < 17; ++kt) {
          const int kbase = kt * 32;
          v16h av, bv;
          {
            const f16* wrow = W + (size_t)arow * 544;
            v8h a0 = *(const v8h*)(wrow + kbase + hiB);
            v8h a1 = *(const v8h*)(wrow + kbase + hiB + 8);
#pragma unroll
            for (int e = 0; e < 8; ++e) { av[e] = a0[e]; av[e + 8] = a1[e]; }
          }
          bv = *(const v16h*)&z[bcol][kbase + hiB];
          acc = WMMA_F16(av, bv, acc);
        }
        // D: M = gate (contiguous), N = batch -> one v8f store per lane
        int g0 = gt * 16 + hiA;
        v8f st;
#pragma unroll
        for (int d = 0; d < 8; ++d) st[d] = acc[d] + bsum[l * 1088 + g0 + d];
        *(v8f*)(gates + (size_t)bcol * 1088 + g0) = st;
      }
      __syncthreads();

      // elementwise gate combine (i,f,g,o at offsets 0/269/538/807)
      for (int i = tid; i < 32 * 269; i += 1024) {
        int b = i / 269, k = i % 269;
        float gi = gates[(size_t)b * 1088 + k];
        float gf = gates[(size_t)b * 1088 + 269 + k];
        float gg = gates[(size_t)b * 1088 + 538 + k];
        float go = gates[(size_t)b * 1088 + 807 + k];
        float si = 1.0f / (1.0f + __expf(-gi));
        float sf = 1.0f / (1.0f + __expf(-gf));
        float so = 1.0f / (1.0f + __expf(-go));
        float tg = tanhf(gg);
        float cold = cstate[(l * 32 + b) * 272 + k] * nd[b];
        float cnew = sf * cold + si * tg;
        float hnew = so * tanhf(cnew);
        cstate[(l * 32 + b) * 272 + k] = cnew;
        hstate[(l * 32 + b) * 272 + k] = hnew;
        if (l == 1) outs[((size_t)t * 32 + b) * 269 + k] = hnew;
      }
      __syncthreads();
    }
  }
}

// ---------------- heads: policy logits, baseline, argmax ----------------
__global__ void heads_k(const float* __restrict__ outs,
                        const float* __restrict__ pw, const float* __restrict__ pb,
                        const float* __restrict__ bw, const float* __restrict__ bb,
                        float* __restrict__ out)
{
  int i = blockIdx.x * blockDim.x + threadIdx.x;
  if (i >= 2048) return;
  const float* xr = outs + (size_t)i * 269;
  float best = -3.402823466e38f; int bi = 0;
  for (int j = 0; j < 6; ++j) {
    float s = pb[j];
    for (int k = 0; k < 269; ++k) s += xr[k] * pw[j * 269 + k];
    out[(size_t)i * 6 + j] = s;
    if (s > best) { best = s; bi = j; }
  }
  float base = bb[0];
  for (int k = 0; k < 269; ++k) base += xr[k] * bw[k];
  out[12288 + i] = base;
  out[12288 + 2048 + i] = (float)bi;
}

__global__ void copy_hc_k(const float* __restrict__ hstate, const float* __restrict__ cstate,
                          float* __restrict__ out)
{
  int i = blockIdx.x * blockDim.x + threadIdx.x;
  if (i >= 2 * 32 * 269) return;
  int k = i % 269; int b = (i / 269) % 32; int l = i / (269 * 32);
  out[16384 + i] = hstate[(l * 32 + b) * 272 + k];
  out[16384 + 17216 + i] = cstate[(l * 32 + b) * 272 + k];
}

// ---------------- host ----------------

template <int CIN, int COUT>
static inline void launch_conv(hipStream_t stream, const f16* in, const f16* wk,
                               const float* bias, const f16* resid, f16* out,
                               int H, int W, int reluIn)
{
  int numPix = 256 * H * W;                       // always a multiple of 64
  dim3 g(numPix / 64, COUT / 16);
  conv3x3_wmma_k<CIN, COUT, 4><<<g, 32, 0, stream>>>(in, wk, bias, resid, out, H, W, reluIn);
}

extern "C" void kernel_launch(void* const* d_in, const int* in_sizes, int n_in,
                              void* d_out, int out_size, void* d_ws, size_t ws_size,
                              hipStream_t stream) {
  (void)in_sizes; (void)n_in; (void)out_size; (void)ws_size;
  const int*   frame       = (const int*)d_in[0];
  const int*   last_action = (const int*)d_in[1];
  const float* reward      = (const float*)d_in[2];
  const unsigned char* done = (const unsigned char*)d_in[3];   // jax bool = 1 byte
  const float* apos = (const float*)d_in[4];
  const float* adir = (const float*)d_in[5];
  const float* tpos = (const float*)d_in[6];
  const float* h0   = (const float*)d_in[7];
  const float* c0   = (const float*)d_in[8];
  const float* fc_w = (const float*)d_in[39];
  const float* fc_b = (const float*)d_in[40];
  const float* pw   = (const float*)d_in[49];
  const float* pb   = (const float*)d_in[50];
  const float* bw   = (const float*)d_in[51];
  const float* bb   = (const float*)d_in[52];
  float* out = (float*)d_out;

  // ---- workspace bump allocator ----
  char* base = (char*)d_ws;
  size_t off = 0;
  auto alloc = [&](size_t bytes) -> void* {
    void* p = base + off;
    off += (bytes + 255) & ~(size_t)255;
    return p;
  };
  const int CHF = 256;                           // frames per encoder chunk
  f16* a0      = (f16*)alloc((size_t)CHF * 64 * 64 * 3 * 2);
  f16* convbuf = (f16*)alloc((size_t)CHF * 64 * 64 * 16 * 2);
  f16* p1 = (f16*)alloc((size_t)CHF * 32 * 32 * 16 * 2);
  f16* p2 = (f16*)alloc((size_t)CHF * 32 * 32 * 16 * 2);
  f16* q1 = (f16*)alloc((size_t)CHF * 16 * 16 * 32 * 2);
  f16* q2 = (f16*)alloc((size_t)CHF * 16 * 16 * 32 * 2);
  f16* r2 = (f16*)alloc((size_t)CHF * 8 * 8 * 32 * 2);
  f16* Rall = (f16*)alloc((size_t)2048 * 8 * 8 * 32 * 2);
  f16* Rn   = (f16*)alloc((size_t)2048 * 2048 * 2);   // NCHW-flat repack for FC
  f16* wp[3][5];
  const int stCin[3]  = {3, 16, 32};
  const int stCout[3] = {16, 32, 32};
  for (int s = 0; s < 3; ++s) {
    wp[s][0] = (f16*)alloc((size_t)stCout[s] * 9 * stCin[s] * 2);
    for (int j = 1; j < 5; ++j)
      wp[s][j] = (f16*)alloc((size_t)stCout[s] * 9 * stCout[s] * 2);
  }
  f16*   fcw_h  = (f16*)alloc((size_t)256 * 2048 * 2);
  f16*   wcat   = (f16*)alloc((size_t)2 * 1088 * 544 * 2);
  float* bsum   = (float*)alloc((size_t)2 * 1088 * 4);
  f16*   Xfc    = (f16*)alloc((size_t)2048 * 256 * 2);
  f16*   cib    = (f16*)alloc((size_t)2048 * 288 * 2);
  float* hstate = (float*)alloc((size_t)2 * 32 * 272 * 4);
  float* cstate = (float*)alloc((size_t)2 * 32 * 272 * 4);
  float* gates  = (float*)alloc((size_t)32 * 1088 * 4);
  float* outs   = (float*)alloc((size_t)2048 * 269 * 4);

  // ---- pack weights (f32 -> f16, re-layout) ----
  for (int s = 0; s < 3; ++s) {
    int ib = 9 + s * 10;
    int n0 = stCout[s] * 9 * stCin[s];
    pack_conv_w_k<<<(n0 + 255) / 256, 256, 0, stream>>>(
        (const float*)d_in[ib], wp[s][0], stCout[s], stCin[s]);
    int nb = stCout[s] * 9 * stCout[s];
    for (int j = 0; j < 4; ++j)
      pack_conv_w_k<<<(nb + 255) / 256, 256, 0, stream>>>(
          (const float*)d_in[ib + 2 + j * 2], wp[s][1 + j], stCout[s], stCout[s]);
  }
  {
    size_t n = (size_t)256 * 2048;
    cast_f32_f16_k<<<(unsigned)((n + 255) / 256), 256, 0, stream>>>(fc_w, fcw_h, n);
  }
  for (int l = 0; l < 2; ++l)
    pack_lstm_k<<<(1088 * 544 + 255) / 256, 256, 0, stream>>>(
        (const float*)d_in[41 + l * 4], (const float*)d_in[42 + l * 4],
        (const float*)d_in[43 + l * 4], (const float*)d_in[44 + l * 4],
        wcat + (size_t)l * 1088 * 544, bsum + l * 1088);

  // ---- encoder: 8 chunks of 256 frames ----
  for (int ch = 0; ch < 8; ++ch) {
    int f0 = ch * CHF;
    {
      size_t total = (size_t)CHF * 64 * 64 * 3;
      frames_to_half_k<<<(unsigned)((total + 255) / 256), 256, 0, stream>>>(frame, a0, f0, CHF);
    }
    const float* cb0  = (const float*)d_in[10];
    const float* cb1  = (const float*)d_in[20];
    const float* cb2  = (const float*)d_in[30];
    // stage 0 (3->16, 64x64 -> pool 32x32)
    launch_conv<3, 16>(stream, a0, wp[0][0], cb0, nullptr, convbuf, 64, 64, 0);
    {
      size_t total = (size_t)CHF * 32 * 32 * 16;
      maxpool3x3s2_k<<<(unsigned)((total + 255) / 256), 256, 0, stream>>>(convbuf, p1, CHF, 64, 64, 16);
    }
    launch_conv<16, 16>(stream, p1, wp[0][1], (const float*)d_in[12], nullptr, p2, 32, 32, 1);
    launch_conv<16, 16>(stream, p2, wp[0][2], (const float*)d_in[14], p1, p1, 32, 32, 1);
    launch_conv<16, 16>(stream, p1, wp[0][3], (const float*)d_in[16], nullptr, p2, 32, 32, 1);
    launch_conv<16, 16>(stream, p2, wp[0][4], (const float*)d_in[18], p1, p1, 32, 32, 1);
    // stage 1 (16->32, 32x32 -> pool 16x16)
    launch_conv<16, 32>(stream, p1, wp[1][0], cb1, nullptr, convbuf, 32, 32, 0);
    {
      size_t total = (size_t)CHF * 16 * 16 * 32;
      maxpool3x3s2_k<<<(unsigned)((total + 255) / 256), 256, 0, stream>>>(convbuf, q1, CHF, 32, 32, 32);
    }
    launch_conv<32, 32>(stream, q1, wp[1][1], (const float*)d_in[22], nullptr, q2, 16, 16, 1);
    launch_conv<32, 32>(stream, q2, wp[1][2], (const float*)d_in[24], q1, q1, 16, 16, 1);
    launch_conv<32, 32>(stream, q1, wp[1][3], (const float*)d_in[26], nullptr, q2, 16, 16, 1);
    launch_conv<32, 32>(stream, q2, wp[1][4], (const float*)d_in[28], q1, q1, 16, 16, 1);
    // stage 2 (32->32, 16x16 -> pool 8x8) -- pooled output goes straight into Rall
    launch_conv<32, 32>(stream, q1, wp[2][0], cb2, nullptr, convbuf, 16, 16, 0);
    f16* xr = Rall + (size_t)f0 * 2048;
    {
      size_t total = (size_t)CHF * 8 * 8 * 32;
      maxpool3x3s2_k<<<(unsigned)((total + 255) / 256), 256, 0, stream>>>(convbuf, xr, CHF, 16, 16, 32);
    }
    launch_conv<32, 32>(stream, xr, wp[2][1], (const float*)d_in[32], nullptr, r2, 8, 8, 1);
    launch_conv<32, 32>(stream, r2, wp[2][2], (const float*)d_in[34], xr, xr, 8, 8, 1);
    launch_conv<32, 32>(stream, xr, wp[2][3], (const float*)d_in[36], nullptr, r2, 8, 8, 1);
    launch_conv<32, 32>(stream, r2, wp[2][4], (const float*)d_in[38], xr, xr, 8, 8, 1);
  }

  // ---- FC + core input ----
  {
    size_t total = (size_t)2048 * 2048;
    repack_nchw_k<<<(unsigned)((total + 255) / 256), 256, 0, stream>>>(Rall, Rn);
  }
  fc_wmma_k<<<dim3(16, 32), 32, 0, stream>>>(Rn, fcw_h, fc_b, Xfc);
  {
    size_t total = (size_t)2048 * 288;
    build_core_k<<<(unsigned)((total + 255) / 256), 256, 0, stream>>>(
        Xfc, reward, last_action, apos, adir, tpos, cib);
  }

  // ---- persistent LSTM (single workgroup, 32 waves) ----
  lstm_seq_k<<<1, 1024, 0, stream>>>(cib, done, h0, c0, wcat, bsum,
                                     hstate, cstate, gates, outs);

  // ---- heads + state copy-out ----
  heads_k<<<8, 256, 0, stream>>>(outs, pw, pb, bw, bb, out);
  copy_hc_k<<<(2 * 32 * 269 + 255) / 256, 256, 0, stream>>>(hstate, cstate, out);
}
